// Attention3D_fusion_50294067036390
// MI455X (gfx1250) — matrix operand order repaired and checked by reading, not compiled, not run on hardware
//
#include <hip/hip_runtime.h>

// ---------------------------------------------------------------------------
// Types for CDNA5 WMMA (wave32): D(16x16,f32) = A(16x32,bf16) x B(32x16,bf16) + C
// ---------------------------------------------------------------------------
typedef __attribute__((ext_vector_type(16))) __bf16 v16bf;
typedef __attribute__((ext_vector_type(8)))  __bf16 bf16x8;
typedef __attribute__((ext_vector_type(4)))  __bf16 bf16x4;
typedef __attribute__((ext_vector_type(8)))  float  v8f;
typedef __attribute__((ext_vector_type(4)))  float  f32x4;

union V16 { v16bf v; bf16x8 h[2]; };

constexpr int   kB = 16, kN = 1024, kC = 512, kH = 8, kD = 64;
constexpr float kScale = 0.125f;   // 64^-0.5

static __device__ __forceinline__ v8f wmma_bf16(v16bf a, v16bf b, v8f c) {
  // emits v_wmma_f32_16x16x32_bf16
  return __builtin_amdgcn_wmma_f32_16x16x32_bf16(false, a, false, b, (short)0, c,
                                                 false, false);
}

// LDS byte offset of a generic pointer to __shared__ data
static __device__ __forceinline__ unsigned lds_off(const void* p) {
  return (unsigned)(unsigned long long)
         (__attribute__((address_space(3))) const void*)p;
}

// Async DMA: 16B global -> LDS, tracked by ASYNCcnt (no VGPR data movement)
static __device__ __forceinline__ void async_b128(unsigned ldsOff,
                                                  const __bf16* g) {
  asm volatile("global_load_async_to_lds_b128 %0, %1, off"
               :: "v"(ldsOff), "v"(g) : "memory");
}

static __device__ __forceinline__ void wait_async0() {
  asm volatile("s_wait_asynccnt 0x0" ::: "memory");
}

// ---------------------------------------------------------------------------
// fp32 -> bf16 convert (weights)
// ---------------------------------------------------------------------------
__global__ void __launch_bounds__(256) f2bf_kernel(const float* __restrict__ in,
                                                   __bf16* __restrict__ out, int n) {
  int i = (blockIdx.x * 256 + threadIdx.x) * 4;
  if (i < n) {
    f32x4 v = *(const f32x4*)(in + i);
    bf16x4 o;
#pragma unroll
    for (int j = 0; j < 4; ++j) o[j] = (__bf16)v[j];
    *(bf16x4*)(out + i) = o;
  }
}

// ---------------------------------------------------------------------------
// QKV projection.  One wave -> 16 rows x 64 cols (= one head's D range).
// z=0: Q = x2 @ Wq^T  -> row-major [B*H, N, D]
// z=1: K = x  @ Wk^T  -> row-major [B*H, N, D]
// z=2: V = x  @ Wv^T  -> TRANSPOSED [B*H, D, N]  (direct accumulator store)
// ---------------------------------------------------------------------------
__global__ void __launch_bounds__(128)
qkv_gemm(const float* __restrict__ x, const float* __restrict__ x2,
         const __bf16* __restrict__ Wq, const __bf16* __restrict__ Wk,
         const __bf16* __restrict__ Wv,
         __bf16* __restrict__ Qo, __bf16* __restrict__ Ko, __bf16* __restrict__ Vt) {
  const int z    = blockIdx.z;
  const float*  src = (z == 0) ? x2 : x;
  const __bf16* W   = (z == 0) ? Wq : (z == 1 ? Wk : Wv);

  const int wv   = threadIdx.x >> 5;
  const int lane = threadIdx.x & 31;
  const int lrow = lane & 15;      // row (A/B operand) or column (C/D) index
  const int hf   = lane >> 4;      // half-wave select

  const int m0 = (blockIdx.x * 4 + wv) * 16;   // row tile in [B*N]
  const int n0 = blockIdx.y * 64;              // 64-wide strip == one head

  const float* arow = src + (size_t)(m0 + lrow) * kC;

  v8f acc[4] = {};                 // 4 n-tiles of 16
  for (int k0 = 0; k0 < kC; k0 += 32) {
    // A operand: lane holds row (lane&15); chunk0 k = k0+hf*8.., chunk1 = +16
    V16 a;
    f32x4 f0 = *(const f32x4*)(arow + k0 + hf * 8);
    f32x4 f1 = *(const f32x4*)(arow + k0 + hf * 8 + 4);
    f32x4 f2 = *(const f32x4*)(arow + k0 + 16 + hf * 8);
    f32x4 f3 = *(const f32x4*)(arow + k0 + 16 + hf * 8 + 4);
#pragma unroll
    for (int j = 0; j < 4; ++j) {
      a.v[j]      = (__bf16)f0[j];
      a.v[4 + j]  = (__bf16)f1[j];
      a.v[8 + j]  = (__bf16)f2[j];
      a.v[12 + j] = (__bf16)f3[j];
    }
#pragma unroll
    for (int t = 0; t < 4; ++t) {
      // B operand column n = W row n (Y = X @ W^T): chunk0 k = k0+hf*8.., chunk1 = +16
      const __bf16* wrow = W + (size_t)(n0 + t * 16 + lrow) * kC + k0 + hf * 8;
      V16 b;
      b.h[0] = *(const bf16x8*)(wrow);
      b.h[1] = *(const bf16x8*)(wrow + 16);
      acc[t] = wmma_bf16(a.v, b.v, acc[t]);
    }
  }

  const int b  = m0 >> 10;         // / kN
  const int nn = m0 & (kN - 1);
  const int h  = n0 >> 6;

  if (z < 2) {
    // LDS transpose -> coalesced row-major 16B stores
    __shared__ __attribute__((aligned(16))) __bf16 lds[4][16 * 72];
    __bf16* tb = lds[wv];
#pragma unroll
    for (int t = 0; t < 4; ++t)
#pragma unroll
      for (int v = 0; v < 8; ++v)
        tb[(v + hf * 8) * 72 + t * 16 + lrow] = (__bf16)acc[t][v];
    asm volatile("" ::: "memory");           // wave-private LDS, DS ops in-order
    const int r = lane >> 1, seg = lane & 1;
    __bf16* dst = (z == 0 ? Qo : Ko) +
                  ((size_t)(b * kH + h) * kN + nn + r) * kD + seg * 32;
    const bf16x8* sv = (const bf16x8*)&tb[r * 72 + seg * 32];
#pragma unroll
    for (int i = 0; i < 4; ++i) *(bf16x8*)(dst + i * 8) = sv[i];
  } else {
    // Transposed V store: accumulator VGPR index == row -> contiguous in N
    __bf16* dst = Vt + (size_t)(b * kH + h) * kD * kN;
#pragma unroll
    for (int t = 0; t < 4; ++t) {
      bf16x8 p;
#pragma unroll
      for (int v = 0; v < 8; ++v) p[v] = (__bf16)acc[t][v];
      *(bf16x8*)(dst + (size_t)(t * 16 + lrow) * kN + nn + hf * 8) = p;
    }
  }
}

// ---------------------------------------------------------------------------
// Flash cross-attention, LDS-staged.
// Block = 8 waves = 256 queries of one (b,h). Each wave owns 32 queries.
// K/V chunks (32 keys) are staged cooperatively into double-buffered LDS with
// GLOBAL_LOAD_ASYNC_TO_LDS_B128 (ASYNCcnt) and consumed via ds_load_b128.
// Q,K: [B*H, N, D] bf16,  Vt: [B*H, D, N] bf16,  Ao: [B*N, C] bf16
// ---------------------------------------------------------------------------
__global__ void __launch_bounds__(256)
attn_kernel(const __bf16* __restrict__ Q, const __bf16* __restrict__ K,
            const __bf16* __restrict__ Vt, __bf16* __restrict__ Ao) {
  // per buffer: K chunk [32 keys][64 d] = 2048 elems, then Vt chunk [64 d][32 keys]
  __shared__ __attribute__((aligned(16))) __bf16 kv[2][4096];
  __shared__ __attribute__((aligned(16))) __bf16 pbuf[8][16 * 72];

  const int tid  = threadIdx.x;
  const int wv   = tid >> 5;
  const int lane = tid & 31;
  const int lrow = lane & 15;
  const int hf   = lane >> 4;

  const int bh = blockIdx.y;
  const int q0 = (blockIdx.x * 8 + wv) * 32;   // 32 queries per wave

  // Q tiles as A operands (two 16-row tiles x two K=32 steps), loaded once.
  V16 qa[2][2];
#pragma unroll
  for (int qt = 0; qt < 2; ++qt) {
    const __bf16* qrow = Q + ((size_t)bh * kN + q0 + qt * 16 + lrow) * kD;
#pragma unroll
    for (int kt = 0; kt < 2; ++kt) {
      qa[qt][kt].h[0] = *(const bf16x8*)(qrow + kt * 32 + hf * 8);
      qa[qt][kt].h[1] = *(const bf16x8*)(qrow + kt * 32 + 16 + hf * 8);
    }
  }

  float mi[2][8], li[2][8];
#pragma unroll
  for (int qt = 0; qt < 2; ++qt)
#pragma unroll
    for (int v = 0; v < 8; ++v) { mi[qt][v] = -3.0e38f; li[qt][v] = 0.0f; }
  v8f o[2][4] = {};

  const __bf16* kg = K + (size_t)bh * kN * kD;
  const __bf16* vg = Vt + (size_t)bh * kD * kN;
  __bf16* P = pbuf[wv];

  // cooperative async stage of one 32-key chunk: 256 thr x 2 x 16B = 8KB
  auto stage = [&](int buf, int j0) {
    async_b128(lds_off(&kv[buf][tid * 8]),
               kg + (size_t)j0 * kD + tid * 8);
    async_b128(lds_off(&kv[buf][2048 + (tid >> 2) * 32 + (tid & 3) * 8]),
               vg + (size_t)(tid >> 2) * kN + j0 + (tid & 3) * 8);
  };

  stage(0, 0);
  wait_async0();
  __syncthreads();

  for (int j0 = 0; j0 < kN; j0 += 32) {
    const int cur = (j0 >> 5) & 1;
    if (j0 + 32 < kN) stage(cur ^ 1, j0 + 32);   // prefetch next chunk

    const __bf16* kbuf = kv[cur];          // [32][64]
    const __bf16* vbuf = kv[cur] + 2048;   // [64][32]

#pragma unroll
    for (int qt = 0; qt < 2; ++qt) {
      // ---- S = Q K^T : 16 queries x 32 keys (4 WMMAs, B from LDS)
      v8f s[2] = {};
#pragma unroll
      for (int jt = 0; jt < 2; ++jt) {
#pragma unroll
        for (int kt = 0; kt < 2; ++kt) {
          const __bf16* kr = kbuf + (jt * 16 + lrow) * 64 + kt * 32 + hf * 8;
          V16 bm;
          bm.h[0] = *(const bf16x8*)(kr);
          bm.h[1] = *(const bf16x8*)(kr + 16);
          s[jt] = wmma_bf16(qa[qt][kt].v, bm.v, s[jt]);
        }
      }
      // ---- online softmax (rows live across 16-lane groups)
#pragma unroll
      for (int v = 0; v < 8; ++v) {
        float a0 = s[0][v] * kScale;
        float a1 = s[1][v] * kScale;
        float mx = fmaxf(a0, a1);
#pragma unroll
        for (int off = 1; off < 16; off <<= 1)
          mx = fmaxf(mx, __shfl_xor(mx, off, 16));
        float mnew = fmaxf(mi[qt][v], mx);
        float corr = __expf(mi[qt][v] - mnew);
        mi[qt][v] = mnew;
        float p0 = __expf(a0 - mnew);
        float p1 = __expf(a1 - mnew);
        float rs = p0 + p1;
#pragma unroll
        for (int off = 1; off < 16; off <<= 1) rs += __shfl_xor(rs, off, 16);
        li[qt][v] = li[qt][v] * corr + rs;
#pragma unroll
        for (int t = 0; t < 4; ++t) o[qt][t][v] *= corr;
        // stash P (16x32) row-major in wave-private LDS for layout swap
        P[(v + hf * 8) * 72 + lrow]      = (__bf16)p0;
        P[(v + hf * 8) * 72 + 16 + lrow] = (__bf16)p1;
      }
      asm volatile("" ::: "memory");
      // ---- reload P in A-operand layout (wave-private, DS in-order)
      V16 pa;
      pa.h[0] = *(const bf16x8*)&P[lrow * 72 + hf * 8];
      pa.h[1] = *(const bf16x8*)&P[lrow * 72 + 16 + hf * 8];
      asm volatile("" ::: "memory");
      // ---- O += P V (4 WMMAs, B-operand from staged transposed V in LDS)
#pragma unroll
      for (int t = 0; t < 4; ++t) {
        const __bf16* vr = vbuf + (t * 16 + lrow) * 32 + hf * 8;
        V16 bm;
        bm.h[0] = *(const bf16x8*)(vr);
        bm.h[1] = *(const bf16x8*)(vr + 16);
        o[qt][t] = wmma_bf16(pa.v, bm.v, o[qt][t]);
      }
    }
    // double-buffer handshake: my prefetch landed + everyone done with `cur`
    wait_async0();
    __syncthreads();
  }

  // normalize rows, then LDS transpose -> coalesced bf16 row stores
  const int b = bh >> 3, h = bh & 7;
  const int r = lane >> 1, seg = lane & 1;
#pragma unroll
  for (int qt = 0; qt < 2; ++qt) {
#pragma unroll
    for (int v = 0; v < 8; ++v) {
      float inv = 1.0f / li[qt][v];
#pragma unroll
      for (int t = 0; t < 4; ++t) o[qt][t][v] *= inv;
    }
    __bf16* Ot = pbuf[wv];
#pragma unroll
    for (int t = 0; t < 4; ++t)
#pragma unroll
      for (int v = 0; v < 8; ++v)
        Ot[(v + hf * 8) * 72 + t * 16 + lrow] = (__bf16)o[qt][t][v];
    asm volatile("" ::: "memory");
    __bf16* drow = Ao + ((size_t)(b * kN + q0 + qt * 16 + r)) * kC +
                   h * 64 + seg * 32;
    const bf16x8* sv = (const bf16x8*)&Ot[r * 72 + seg * 32];
#pragma unroll
    for (int i = 0; i < 4; ++i) *(bf16x8*)(drow + i * 8) = sv[i];
    asm volatile("" ::: "memory");   // Ot reused across qt (wave-private)
  }
}

// ---------------------------------------------------------------------------
// Output projection: out = A @ Wp^T + bp  (f32 output, LDS tile transpose)
// ---------------------------------------------------------------------------
__global__ void __launch_bounds__(128)
proj_gemm(const __bf16* __restrict__ A, const __bf16* __restrict__ Wp,
          const float* __restrict__ bp, float* __restrict__ out) {
  const int wv   = threadIdx.x >> 5;
  const int lane = threadIdx.x & 31;
  const int lrow = lane & 15;
  const int hf   = lane >> 4;

  const int m0 = (blockIdx.x * 4 + wv) * 16;
  const int n0 = blockIdx.y * 64;

  const __bf16* arow = A + (size_t)(m0 + lrow) * kC;
  v8f acc[4] = {};
  for (int k0 = 0; k0 < kC; k0 += 32) {
    V16 a;
    a.h[0] = *(const bf16x8*)(arow + k0 + hf * 8);
    a.h[1] = *(const bf16x8*)(arow + k0 + 16 + hf * 8);
#pragma unroll
    for (int t = 0; t < 4; ++t) {
      const __bf16* wrow = Wp + (size_t)(n0 + t * 16 + lrow) * kC + k0 + hf * 8;
      V16 b;
      b.h[0] = *(const bf16x8*)(wrow);
      b.h[1] = *(const bf16x8*)(wrow + 16);
      acc[t] = wmma_bf16(a.v, b.v, acc[t]);
    }
  }
  // bias: accumulator column = n0 + t*16 + (lane&15), uniform over VGPR index
#pragma unroll
  for (int t = 0; t < 4; ++t) {
    float bias = bp[n0 + t * 16 + lrow];
#pragma unroll
    for (int v = 0; v < 8; ++v) acc[t][v] += bias;
  }
  // f32 LDS transpose -> coalesced 16B stores
  __shared__ __attribute__((aligned(16))) float flds[4][16 * 72];
  float* tb = flds[wv];
#pragma unroll
  for (int t = 0; t < 4; ++t)
#pragma unroll
    for (int v = 0; v < 8; ++v)
      tb[(v + hf * 8) * 72 + t * 16 + lrow] = acc[t][v];
  asm volatile("" ::: "memory");
  const int r = lane >> 1, seg = lane & 1;
  float* drow = out + (size_t)(m0 + r) * kC + n0 + seg * 32;
  const f32x4* sv = (const f32x4*)&tb[r * 72 + seg * 32];
#pragma unroll
  for (int i = 0; i < 8; ++i) *(f32x4*)(drow + i * 4) = sv[i];
}

// ---------------------------------------------------------------------------
// Launch
// ---------------------------------------------------------------------------
extern "C" void kernel_launch(void* const* d_in, const int* in_sizes, int n_in,
                              void* d_out, int out_size, void* d_ws, size_t ws_size,
                              hipStream_t stream) {
  (void)in_sizes; (void)n_in; (void)out_size; (void)ws_size;
  const float* x  = (const float*)d_in[0];
  const float* x2 = (const float*)d_in[1];
  const float* Wq = (const float*)d_in[2];
  const float* Wk = (const float*)d_in[3];
  const float* Wv = (const float*)d_in[4];
  const float* Wp = (const float*)d_in[5];
  const float* bp = (const float*)d_in[6];
  float* out = (float*)d_out;

  const size_t nW  = (size_t)kC * kC;          // 262144
  const size_t nT  = (size_t)kB * kN * kC;     // 8388608
  __bf16* wsb = (__bf16*)d_ws;
  __bf16* Wqb = wsb;
  __bf16* Wkb = Wqb + nW;
  __bf16* Wvb = Wkb + nW;
  __bf16* Wpb = Wvb + nW;
  __bf16* Qb  = Wpb + nW;
  __bf16* Kb  = Qb + nT;
  __bf16* Vtb = Kb + nT;
  __bf16* Ab  = Vtb + nT;

  // 1) convert weights to bf16
  {
    int blocks = (int)(nW / (256 * 4));
    f2bf_kernel<<<blocks, 256, 0, stream>>>(Wq, Wqb, (int)nW);
    f2bf_kernel<<<blocks, 256, 0, stream>>>(Wk, Wkb, (int)nW);
    f2bf_kernel<<<blocks, 256, 0, stream>>>(Wv, Wvb, (int)nW);
    f2bf_kernel<<<blocks, 256, 0, stream>>>(Wp, Wpb, (int)nW);
  }
  // 2) Q/K/V projections (z selects matrix; V stored transposed)
  qkv_gemm<<<dim3(kB * kN / 16 / 4, kC / 64, 3), 128, 0, stream>>>(
      x, x2, Wqb, Wkb, Wvb, Qb, Kb, Vtb);
  // 3) fused flash attention (LDS-staged K/V, async copies, double-buffered)
  attn_kernel<<<dim3(kN / 256, kB * kH), 256, 0, stream>>>(Qb, Kb, Vtb, Ab);
  // 4) output projection + bias
  proj_gemm<<<dim3(kB * kN / 16 / 4, kC / 64), 128, 0, stream>>>(Ab, Wpb, bp, out);
}
